// AttentionHead_29927332119121
// MI455X (gfx1250) — compile-verified, matrix operand order checked by reading, not run on hardware
//
#include <hip/hip_runtime.h>

// ---------------------------------------------------------------------------
// Causal single-head attention for MI455X (gfx1250, wave32, WMMA bf16).
//   x:[4,2048,1024] f32 ; Wq/Wk/Wv:[1024,128] f32 ; bq/bk/bv:[128] f32
//   out:[4,2048,128] f32
// ---------------------------------------------------------------------------

typedef __attribute__((ext_vector_type(16))) __bf16 v16bf;
typedef __attribute__((ext_vector_type(8)))  float  v8f;

#define WMMA_BF16(A, B, C) \
  __builtin_amdgcn_wmma_f32_16x16x32_bf16(false, (A), false, (B), (short)0, (C), false, false)

constexpr int kB      = 4;
constexpr int kT      = 2048;
constexpr int kNM     = 1024;    // model dim
constexpr int kD      = 128;     // head dim (n_k == n_v)
constexpr int kRows   = kB * kT; // 8192

// ---------------------------------------------------------------------------
// Kernel 0: convert W[1024,128] f32 -> Wt[128,1024] bf16 (transposed), x3 mats
// ---------------------------------------------------------------------------
__global__ void prep_weights(const float* __restrict__ Wq,
                             const float* __restrict__ Wk,
                             const float* __restrict__ Wv,
                             __bf16* __restrict__ Wt) {
  int i = blockIdx.x * blockDim.x + threadIdx.x;
  constexpr int per = kD * kNM;             // 131072
  if (i >= 3 * per) return;
  int mat = i / per;
  int rem = i - mat * per;
  int n = rem / kNM;                        // 0..127
  int k = rem - n * kNM;                    // 0..1023
  const float* W = (mat == 0) ? Wq : (mat == 1) ? Wk : Wv;
  Wt[i] = (__bf16)W[k * kD + n];
}

// ---------------------------------------------------------------------------
// Kernel 1: QKV projection GEMM.  [8192,1024] x [1024,128] (+bias)
//   grid = (8192/64, 128/16, 3), block = 128 (4 waves, 16 rows each)
//   z=0 -> Qb[8192,128] bf16 ; z=1 -> Kb[8192,128] bf16 ; z=2 -> Vt[128,8192] bf16
// ---------------------------------------------------------------------------
__device__ __forceinline__ v16bf cvt16_f32_bf16(const float* __restrict__ p) {
  const float4* q = (const float4*)p;
  float4 f0 = q[0], f1 = q[1], f2 = q[2], f3 = q[3];
  float buf[16] = {f0.x, f0.y, f0.z, f0.w, f1.x, f1.y, f1.z, f1.w,
                   f2.x, f2.y, f2.z, f2.w, f3.x, f3.y, f3.z, f3.w};
  v16bf a;
#pragma unroll
  for (int i = 0; i < 16; ++i) a[i] = (__bf16)buf[i];
  return a;
}

__global__ __launch_bounds__(128) void qkv_gemm(
    const float* __restrict__ x, const __bf16* __restrict__ Wt,
    const float* __restrict__ bq, const float* __restrict__ bk,
    const float* __restrict__ bv,
    __bf16* __restrict__ Qb, __bf16* __restrict__ Kb, __bf16* __restrict__ Vt) {
  const int z    = blockIdx.z;
  const int wave = threadIdx.x >> 5;
  const int lane = threadIdx.x & 31;
  const int hi   = lane >> 4;    // half-wave: 0 or 1
  const int l16  = lane & 15;

  const int m0 = blockIdx.x * 64 + wave * 16;
  const int n0 = blockIdx.y * 16;

  const __bf16* W    = Wt + z * (kD * kNM);
  const float*  bias = (z == 0) ? bq : (z == 1) ? bk : bv;

  v8f acc = {};
  const int arow = (m0 + l16) * kNM;   // A: row m = l16
  const int brow = (n0 + l16) * kNM;   // B: col n = l16 (Wt row)

#pragma unroll 4
  for (int kb = 0; kb < kNM; kb += 32) {
    const int ks = kb + hi * 16;
    v16bf a = cvt16_f32_bf16(x + arow + ks);        // 16 contiguous k of row m
    v16bf b = *(const v16bf*)(W + brow + ks);       // 16 contiguous k of col n
    acc = WMMA_BF16(a, b, acc);
  }

  const float bn = bias[n0 + l16];                  // lane's column bias
  if (z < 2) {
    __bf16* dst = z ? Kb : Qb;
#pragma unroll
    for (int r = 0; r < 8; ++r) {
      const int m = m0 + r + hi * 8;
      dst[m * kD + n0 + l16] = (__bf16)(acc[r] + bn);
    }
  } else {
    // V transposed: Vt[n][m]; lane writes 8 consecutive m's (16B aligned run)
    const int base = (n0 + l16) * kRows + m0 + hi * 8;
#pragma unroll
    for (int r = 0; r < 8; ++r) Vt[base + r] = (__bf16)(acc[r] + bn);
  }
}

// ---------------------------------------------------------------------------
// Kernel 2: causal flash attention, key-split across the 4 waves of a block.
//   grid = (2048/16, 4), block = 128.  Each block owns 16 query rows; wave w
//   processes key tiles j0 = (w + 4i)*32 with private online-softmax state,
//   then the block merges the 4 partials through LDS:
//     out = sum_w e^{m_w-m*} O_w / sum_w e^{m_w-m*} l_w
// ---------------------------------------------------------------------------
__global__ __launch_bounds__(128) void attn_fwd(
    const __bf16* __restrict__ Qb, const __bf16* __restrict__ Kb,
    const __bf16* __restrict__ Vt, float* __restrict__ out) {
  __shared__ alignas(32) __bf16 pbuf[4][16][32];    // per-wave P transpose buffer
  __shared__ float s_o[4][16][kD];                  // per-wave partial O (32 KB)
  __shared__ float s_m[4][16];
  __shared__ float s_l[4][16];

  const int wave = threadIdx.x >> 5;
  const int lane = threadIdx.x & 31;
  const int hi   = lane >> 4;
  const int l16  = lane & 15;

  const int b    = blockIdx.y;
  const int q0   = blockIdx.x * 16;                 // query tile base (within batch)
  const int qend = q0 + 15;
  const float scale = 0.08838834764831845f;         // 1/sqrt(128)

  // Preload Q A-fragments: 4 chunks of k covering d=128 (shared by all waves).
  const int qrow = (b * kT + q0 + l16) * kD;
  v16bf aq[4];
#pragma unroll
  for (int kc = 0; kc < 4; ++kc)
    aq[kc] = *(const v16bf*)(Qb + qrow + kc * 32 + hi * 16);

  v8f o[8];
#pragma unroll
  for (int t = 0; t < 8; ++t) o[t] = (v8f){};
  float mrow[8], lrow[8];
#pragma unroll
  for (int r = 0; r < 8; ++r) { mrow[r] = -__builtin_inff(); lrow[r] = 0.0f; }

  for (int j0 = wave * 32; j0 <= qend; j0 += 128) { // this wave's key tiles
    // Prefetch this wave's next key tile (global_prefetch_b8).
    if (j0 + 128 <= qend)
      __builtin_prefetch(Kb + (b * kT + j0 + 128 + l16) * kD, 0, 1);

    // ---- scores: S[16q x 32k] = Q . K^T ----
    v8f s0 = {}, s1 = {};
    const int krow0 = (b * kT + j0 + l16) * kD;
    const int krow1 = krow0 + 16 * kD;
#pragma unroll
    for (int kc = 0; kc < 4; ++kc) {
      const int ks = kc * 32 + hi * 16;
      v16bf bk0 = *(const v16bf*)(Kb + krow0 + ks);
      v16bf bk1 = *(const v16bf*)(Kb + krow1 + ks);
      s0 = WMMA_BF16(aq[kc], bk0, s0);
      s1 = WMMA_BF16(aq[kc], bk1, s1);
    }

    // ---- causal mask + online softmax (rows live across 16-lane halves) ----
    float p0[8], p1[8], alpha[8];
#pragma unroll
    for (int r = 0; r < 8; ++r) {
      const int qg = q0 + r + hi * 8;
      float v0 = (j0 + l16      <= qg) ? s0[r] * scale : -__builtin_inff();
      float v1 = (j0 + 16 + l16 <= qg) ? s1[r] * scale : -__builtin_inff();
      float tm = fmaxf(v0, v1);
      tm = fmaxf(tm, __shfl_xor(tm, 1, 32));
      tm = fmaxf(tm, __shfl_xor(tm, 2, 32));
      tm = fmaxf(tm, __shfl_xor(tm, 4, 32));
      tm = fmaxf(tm, __shfl_xor(tm, 8, 32));
      const float mn = fmaxf(mrow[r], tm);
      alpha[r] = __expf(mrow[r] - mn);              // exp(-inf)=0 on first tile
      p0[r] = __expf(v0 - mn);
      p1[r] = __expf(v1 - mn);
      float rs = p0[r] + p1[r];
      rs += __shfl_xor(rs, 1, 32);
      rs += __shfl_xor(rs, 2, 32);
      rs += __shfl_xor(rs, 4, 32);
      rs += __shfl_xor(rs, 8, 32);
      lrow[r] = lrow[r] * alpha[r] + rs;
      mrow[r] = mn;
    }

    // ---- transpose P (C-layout f32 -> A-layout bf16) through LDS ----
#pragma unroll
    for (int r = 0; r < 8; ++r) {
      pbuf[wave][r + hi * 8][l16]      = (__bf16)p0[r];
      pbuf[wave][r + hi * 8][16 + l16] = (__bf16)p1[r];
    }
    __asm volatile("s_wait_dscnt 0" ::: "memory");  // wave-private RAW on LDS
    const v16bf pa = *(const v16bf*)&pbuf[wave][l16][hi * 16];

    // ---- O = diag(alpha) * O + P . V  (8 tiles cover d_v = 128) ----
    const int vcol = b * kT + j0 + hi * 16;
#pragma unroll
    for (int t = 0; t < 8; ++t) {
      v16bf vb = *(const v16bf*)(Vt + (t * 16 + l16) * kRows + vcol);
#pragma unroll
      for (int r = 0; r < 8; ++r) o[t][r] *= alpha[r];
      o[t] = WMMA_BF16(pa, vb, o[t]);
    }
  }

  // ---- publish this wave's partial state ----
  if (l16 == 0) {
#pragma unroll
    for (int r = 0; r < 8; ++r) {
      s_m[wave][r + hi * 8] = mrow[r];
      s_l[wave][r + hi * 8] = lrow[r];
    }
  }
#pragma unroll
  for (int t = 0; t < 8; ++t)
#pragma unroll
    for (int r = 0; r < 8; ++r)
      s_o[wave][r + hi * 8][t * 16 + l16] = o[t][r];
  __syncthreads();

  // ---- merge 4 partials: thread tid owns column tid for all 16 rows ----
  const int col = threadIdx.x;                      // 0..127 == d_v
#pragma unroll
  for (int row = 0; row < 16; ++row) {
    const float m0 = s_m[0][row], m1 = s_m[1][row];
    const float m2 = s_m[2][row], m3 = s_m[3][row];
    const float ms = fmaxf(fmaxf(m0, m1), fmaxf(m2, m3));  // finite: wave0 saw key 0
    const float e0 = __expf(m0 - ms), e1 = __expf(m1 - ms);
    const float e2 = __expf(m2 - ms), e3 = __expf(m3 - ms);
    const float denom = e0 * s_l[0][row] + e1 * s_l[1][row] +
                        e2 * s_l[2][row] + e3 * s_l[3][row];
    const float numer = e0 * s_o[0][row][col] + e1 * s_o[1][row][col] +
                        e2 * s_o[2][row][col] + e3 * s_o[3][row][col];
    out[(b * kT + q0 + row) * kD + col] = numer / denom;
  }
}

// ---------------------------------------------------------------------------
// Launcher
// ---------------------------------------------------------------------------
extern "C" void kernel_launch(void* const* d_in, const int* in_sizes, int n_in,
                              void* d_out, int out_size, void* d_ws, size_t ws_size,
                              hipStream_t stream) {
  const float* x  = (const float*)d_in[0];
  const float* Wq = (const float*)d_in[1];
  const float* bq = (const float*)d_in[2];
  const float* Wk = (const float*)d_in[3];
  const float* bk = (const float*)d_in[4];
  const float* Wv = (const float*)d_in[5];
  const float* bv = (const float*)d_in[6];
  float* out = (float*)d_out;

  // Workspace layout (bytes):
  //   [0)        Wt  : 3 * 128*1024 * 2 = 786432
  //   [786432)   Qb  : 8192*128*2       = 2097152
  //   [2883584)  Kb  : 8192*128*2       = 2097152
  //   [4980736)  Vt  : 128*8192*2       = 2097152   (total ~7 MB)
  char* ws = (char*)d_ws;
  __bf16* Wt = (__bf16*)(ws);
  __bf16* Qb = (__bf16*)(ws + 786432);
  __bf16* Kb = (__bf16*)(ws + 786432 + 2097152);
  __bf16* Vt = (__bf16*)(ws + 786432 + 2 * 2097152);

  {
    const int total = 3 * kD * kNM;
    prep_weights<<<(total + 255) / 256, 256, 0, stream>>>(Wq, Wk, Wv, Wt);
  }
  {
    dim3 grid(kRows / 64, kD / 16, 3);   // (128, 8, 3)
    qkv_gemm<<<grid, 128, 0, stream>>>(x, Wt, bq, bk, bv, Qb, Kb, Vt);
  }
  {
    dim3 grid(kT / 16, kB);              // (128, 4) blocks, 4 waves each
    attn_fwd<<<grid, 128, 0, stream>>>(Qb, Kb, Vt, out);
  }
}
